// PytorchFalconDecoderModel_16578573762876
// MI455X (gfx1250) — matrix operand order repaired and checked by reading, not compiled, not run on hardware
//
#include <hip/hip_runtime.h>
#include <hip/hip_bf16.h>

// ---------------------------------------------------------------------------
// Falcon decoder layer (B=1, S=2048, D=2048, H=32, HD=64, FF=8192), fp32 I/O.
// GEMMs on v_wmma_f32_16x16x32_bf16 (bf16 in, f32 accum).  Weights are
// transpose-converted to bf16 once per launch; activations flow as bf16, so
// GEMM tile staging is a pure copy done by the Tensor Data Mover with
// hardware LDS padding, double-buffered so the next tile's DMA overlaps the
// current tile's WMMAs.  Manual-copy fallback if the builtin is unavailable.
// ---------------------------------------------------------------------------

typedef __bf16 bf16;
typedef bf16  v16bf __attribute__((ext_vector_type(16)));
typedef float v8f   __attribute__((ext_vector_type(8)));
typedef unsigned int v4u __attribute__((ext_vector_type(4)));
typedef int  v8i __attribute__((ext_vector_type(8)));
typedef int  v4i __attribute__((ext_vector_type(4)));

#define S_LEN   2048
#define D_DIM   2048
#define NHEAD   32
#define HDIM    64
#define FF_DIM  8192
#define QKV_LD  (3 * D_DIM)

#if defined(__HIP_DEVICE_COMPILE__) && __has_builtin(__builtin_amdgcn_tensor_load_to_lds) && __has_builtin(__builtin_amdgcn_s_wait_tensorcnt)
#define HAVE_TDM 1
#else
#define HAVE_TDM 0
#endif

static __device__ __forceinline__ v8f v8f_zero() {
    v8f z = {0.f, 0.f, 0.f, 0.f, 0.f, 0.f, 0.f, 0.f};
    return z;
}

// Load a WMMA 16-element bf16 fragment from two 16-byte-aligned halves.
static __device__ __forceinline__ v16bf ld_frag(const bf16* p0, const bf16* p1) {
    union { uint4 q[2]; v16bf v; } u;
    u.q[0] = *(const uint4*)p0;
    u.q[1] = *(const uint4*)p1;
    return u.v;
}

#if HAVE_TDM
// Issue a TDM 2D tile load (elements are 2 bytes).  LDS padding: +16B after
// every 64B row => LDS row stride = 40 bf16.  6-arg builtin form.
static __device__ __forceinline__ void tdm_load_tile_2d(
    unsigned lds_off, const bf16* gaddr,
    unsigned tensor_d0, unsigned tensor_d1,
    unsigned tile_d0, unsigned tile_d1, unsigned stride0_elems) {
    unsigned long long ga = (unsigned long long)(uintptr_t)gaddr;
    v4u g0;
    g0[0] = 1u;                                   // count=1, user descriptor
    g0[1] = lds_off;                              // lds_addr
    g0[2] = (unsigned)ga;                         // global_addr[31:0]
    g0[3] = (unsigned)(ga >> 32) | (2u << 30);    // global_addr[56:32], type=2
    v8i g1;
    g1[0] = (int)((1u << 16)        // data_size = 2 bytes
                | (1u << 20)        // pad_enable
                | (3u << 22)        // pad_interval: 16 DWORDs (64B = one row)
                | (3u << 25));      // pad_amount:   4 DWORDs (16B)
    g1[1] = (int)((tensor_d0 & 0xffffu) << 16);                    // dim0[15:0]
    g1[2] = (int)((tensor_d0 >> 16) | ((tensor_d1 & 0xffffu) << 16));
    g1[3] = (int)((tensor_d1 >> 16) | (tile_d0 << 16));
    g1[4] = (int)tile_d1;                                          // tile_dim1 (tile_dim2=0)
    g1[5] = (int)stride0_elems;                                    // dim0 stride [31:0]
    g1[6] = 0;
    g1[7] = 0;
    v4i g2 = {0, 0, 0, 0};
    v4i g3 = {0, 0, 0, 0};
    v8i g4 = {0, 0, 0, 0, 0, 0, 0, 0};
    __builtin_amdgcn_tensor_load_to_lds(g0, g1, g2, g3, g4, 0);
}
#endif

// ---------------------------------------------------------------------------
// LayerNorm -> bf16 output.  One row per block.
// ---------------------------------------------------------------------------
__global__ __launch_bounds__(256)
void ln_kernel(const float* __restrict__ x, const float* __restrict__ g,
               const float* __restrict__ b, bf16* __restrict__ out, int d) {
    const int row = blockIdx.x;
    const float* xr = x + (size_t)row * d;
    float sum = 0.f, sq = 0.f;
    for (int i = threadIdx.x; i < d; i += blockDim.x) {
        float v = xr[i];
        sum += v;
        sq  += v * v;
    }
    __shared__ float s1[256], s2[256];
    s1[threadIdx.x] = sum;
    s2[threadIdx.x] = sq;
    __syncthreads();
    for (int off = 128; off > 0; off >>= 1) {
        if ((int)threadIdx.x < off) {
            s1[threadIdx.x] += s1[threadIdx.x + off];
            s2[threadIdx.x] += s2[threadIdx.x + off];
        }
        __syncthreads();
    }
    const float mean = s1[0] / (float)d;
    const float var  = s2[0] / (float)d - mean * mean;
    const float inv  = rsqrtf(var + 1e-5f);
    for (int i = threadIdx.x; i < d; i += blockDim.x) {
        out[(size_t)row * d + i] = (bf16)((xr[i] - mean) * inv * g[i] + b[i]);
    }
}

// ---------------------------------------------------------------------------
// Transpose + convert: in f32 [R,C] -> out bf16 [C,R].  (weights, once/launch)
// ---------------------------------------------------------------------------
__global__ __launch_bounds__(256)
void transpose_f32_bf16(const float* __restrict__ in, bf16* __restrict__ out,
                        int R, int C) {
    __shared__ float tile[32][33];
    const int c0 = blockIdx.x * 32, r0 = blockIdx.y * 32;
    const int tx = threadIdx.x, ty = threadIdx.y;
    for (int i = ty; i < 32; i += 8)
        tile[i][tx] = in[(size_t)(r0 + i) * C + c0 + tx];
    __syncthreads();
    for (int i = ty; i < 32; i += 8)
        out[(size_t)(c0 + i) * R + r0 + tx] = (bf16)tile[tx][i];
}

// ---------------------------------------------------------------------------
// Per-head V transpose: qkv bf16 [S, 3D] (V slice) -> Vt bf16 [H*HD, S].
// ---------------------------------------------------------------------------
__global__ __launch_bounds__(256)
void vtrans_kernel(const bf16* __restrict__ qkv, bf16* __restrict__ vt) {
    __shared__ bf16 tile[32][33];
    const int head = blockIdx.y >> 1;
    const int hb   = (blockIdx.y & 1) * 32;
    const int s0   = blockIdx.x * 32;
    const int tx = threadIdx.x, ty = threadIdx.y;
    const bf16* src = qkv + 2 * D_DIM + head * HDIM + hb;
    for (int i = ty; i < 32; i += 8)
        tile[i][tx] = src[(size_t)(s0 + i) * QKV_LD + tx];
    __syncthreads();
    for (int i = ty; i < 32; i += 8)
        vt[(size_t)(head * HDIM + hb + i) * S_LEN + s0 + tx] = tile[tx][i];
}

// ---------------------------------------------------------------------------
// WMMA GEMM:  C[M,N] = act(A[M,K] @ Bt[N,K]^T + bias) (+ residual)
// A, Bt bf16 row-major.  Block tile 128x64, K-step 32, 8 waves, each wave
// 32x32 (2x2 wmma).  M%128==0, N%64==0, K%64==0 (true for all calls).
// Two LDS stages: TDM prefetches tile k+1 while tile k is computed
// (TENSORcnt completes in-order; s_wait_tensorcnt 2 releases the current
// stage while the next stage's two loads are still in flight).
// ACT: 0 none, 1 exact GELU.  RES: add f32 residual.  WF/WB: write f32/bf16.
// ---------------------------------------------------------------------------
#define GBM 128
#define GBN 64
#define GBK 32
#define GLDS 40   // padded LDS row stride in bf16 elements (64B + 16B pad)

template <int ACT, bool RES, bool WF, bool WB>
__global__ __launch_bounds__(256)
void gemm_bf16_wmma(const bf16* __restrict__ A, const bf16* __restrict__ Bt,
                    const float* __restrict__ bias,
                    const float* __restrict__ residual,
                    float* __restrict__ Cf, bf16* __restrict__ Cb,
                    int N, int K) {
    __shared__ __attribute__((aligned(16))) bf16 As[2][GBM * GLDS];
    __shared__ __attribute__((aligned(16))) bf16 Bs[2][GBN * GLDS];

    const int tid  = threadIdx.x;
    const int lane = tid & 31;
    const int wid  = tid >> 5;
    const int wm   = (wid >> 1) * 32;
    const int wn   = (wid & 1) * 32;
    const int bm   = blockIdx.y * GBM;
    const int bn   = blockIdx.x * GBN;
    const int lg   = lane >> 4;
    const int ll   = lane & 15;

    v8f acc[2][2];
    acc[0][0] = v8f_zero(); acc[0][1] = v8f_zero();
    acc[1][0] = v8f_zero(); acc[1][1] = v8f_zero();

    const bf16* Ag = A + (size_t)bm * K;
    const bf16* Bg = Bt + (size_t)bn * K;

#if HAVE_TDM
    unsigned as_off[2], bs_off[2];
    as_off[0] = (unsigned)(uintptr_t)(void*)&As[0][0];
    as_off[1] = (unsigned)(uintptr_t)(void*)&As[1][0];
    bs_off[0] = (unsigned)(uintptr_t)(void*)&Bs[0][0];
    bs_off[1] = (unsigned)(uintptr_t)(void*)&Bs[1][0];
    // Prologue: start stage 0.
    if (wid == 0) {
        tdm_load_tile_2d(as_off[0], Ag, (unsigned)K, GBM, GBK, GBM, (unsigned)K);
        tdm_load_tile_2d(bs_off[0], Bg, (unsigned)K, GBN, GBK, GBN, (unsigned)K);
    }
#endif

    int ib = 0;
    for (int k0 = 0; k0 < K; k0 += GBK, ib ^= 1) {
#if HAVE_TDM
        if (wid == 0) {
            if (k0 + GBK < K) {
                // Prefetch next tile into the other stage, then release the
                // current stage (two oldest TDM ops) without draining.
                tdm_load_tile_2d(as_off[ib ^ 1], Ag + k0 + GBK,
                                 (unsigned)K, GBM, GBK, GBM, (unsigned)K);
                tdm_load_tile_2d(bs_off[ib ^ 1], Bg + k0 + GBK,
                                 (unsigned)K, GBN, GBK, GBN, (unsigned)K);
                __builtin_amdgcn_s_wait_tensorcnt(2);
            } else {
                __builtin_amdgcn_s_wait_tensorcnt(0);
            }
        }
        __syncthreads();
#else
        for (int p = tid; p < GBM * GBK / 2; p += 256) {
            int r = p >> 4, c = (p & 15) * 2;
            *(unsigned*)&As[ib][r * GLDS + c] =
                *(const unsigned*)&Ag[(size_t)r * K + k0 + c];
        }
        for (int p = tid; p < GBN * GBK / 2; p += 256) {
            int r = p >> 4, c = (p & 15) * 2;
            *(unsigned*)&Bs[ib][r * GLDS + c] =
                *(const unsigned*)&Bg[(size_t)r * K + k0 + c];
        }
        __syncthreads();
#endif
        // A fragments: lane -> M row; halves at K {lg*8..} and {16+lg*8..}.
        v16bf afr[2], bfr[2];
#pragma unroll
        for (int mt = 0; mt < 2; ++mt) {
            const bf16* ap = &As[ib][(wm + mt * 16 + ll) * GLDS];
            afr[mt] = ld_frag(ap + lg * 8, ap + 16 + lg * 8);
        }
        // B fragments: lane -> N col; halves at K {lg*8..} and {16+lg*8..}.
#pragma unroll
        for (int nt = 0; nt < 2; ++nt) {
            const bf16* bp = &Bs[ib][(wn + nt * 16 + ll) * GLDS];
            bfr[nt] = ld_frag(bp + lg * 8, bp + 16 + lg * 8);
        }
#pragma unroll
        for (int mt = 0; mt < 2; ++mt)
#pragma unroll
            for (int nt = 0; nt < 2; ++nt)
                acc[mt][nt] = __builtin_amdgcn_wmma_f32_16x16x32_bf16(
                    false, afr[mt], false, bfr[nt], (short)0, acc[mt][nt],
                    false, false);
        __syncthreads();
    }

    // Epilogue: C/D layout lane -> N, vgpr r -> M row (r or 8+r).
#pragma unroll
    for (int mt = 0; mt < 2; ++mt)
#pragma unroll
        for (int nt = 0; nt < 2; ++nt)
#pragma unroll
            for (int r = 0; r < 8; ++r) {
                const int row = bm + wm + mt * 16 + lg * 8 + r;
                const int col = bn + wn + nt * 16 + ll;
                float v = acc[mt][nt][r] + bias[col];
                if (ACT == 1) v = 0.5f * v * (1.0f + erff(v * 0.70710678118654752f));
                const size_t idx = (size_t)row * N + col;
                if (RES) v += residual[idx];
                if (WF) Cf[idx] = v;
                if (WB) Cb[idx] = (bf16)v;
            }
}

// ---------------------------------------------------------------------------
// Flash-style causal ALiBi attention.  4 independent waves per block, each
// wave owns a 16-row query block.  qkv bf16 [S,3D]; Vt bf16 [H*HD, S];
// ctx bf16 [S,D].  logits = (q.k + slope*kpos)/8 + causal mask.
// ---------------------------------------------------------------------------
__global__ __launch_bounds__(128)
void attn_kernel(const bf16* __restrict__ qkv, const bf16* __restrict__ vt,
                 bf16* __restrict__ ctx) {
    const int head = blockIdx.y;
    const int tid  = threadIdx.x;
    const int lane = tid & 31;
    const int wid  = tid >> 5;
    const int q0   = blockIdx.x * 64 + wid * 16;
    const int lg   = lane >> 4;
    const int ll   = lane & 15;

    const float slope    = exp2f(-8.0f * (float)(head + 1) / (float)NHEAD);
    const float inv_norm = 0.125f;  // 1/sqrt(64)

    const bf16* Qp = qkv + head * HDIM;
    const bf16* Kp = qkv + D_DIM + head * HDIM;
    const bf16* Vp = vt + (size_t)head * HDIM * S_LEN;

    __shared__ __attribute__((aligned(16))) bf16 Ps[4][16 * GLDS];
    bf16* ps = &Ps[wid][0];

    // Q fragments: 16x64 as two K=32 chunks (A layout), contiguous b128 loads.
    v16bf qfr[2];
    {
        const bf16* qp = Qp + (size_t)(q0 + ll) * QKV_LD;
#pragma unroll
        for (int c = 0; c < 2; ++c)
            qfr[c] = ld_frag(qp + c * 32 + lg * 8, qp + c * 32 + 16 + lg * 8);
    }

    v8f o[4];
#pragma unroll
    for (int c = 0; c < 4; ++c) o[c] = v8f_zero();
    float mrow[8], lrow[8];
#pragma unroll
    for (int r = 0; r < 8; ++r) { mrow[r] = -1e30f; lrow[r] = 0.f; }

    float st[2][8];

    for (int kb = 0; kb <= q0 + 15; kb += 32) {
        // ---- scores: two 16x16 tiles over 32 keys ----
#pragma unroll
        for (int t = 0; t < 2; ++t) {
            v8f s = v8f_zero();
            const int kbase = kb + t * 16;
#pragma unroll
            for (int c = 0; c < 2; ++c) {
                const bf16* kp = Kp + (size_t)(kbase + ll) * QKV_LD + c * 32 + lg * 8;
                v16bf bf_ = ld_frag(kp, kp + 16);
                s = __builtin_amdgcn_wmma_f32_16x16x32_bf16(
                    false, qfr[c], false, bf_, (short)0, s, false, false);
            }
            const int kpos = kbase + ll;
#pragma unroll
            for (int r = 0; r < 8; ++r) {
                const int qrow = q0 + lg * 8 + r;
                float v = (s[r] + slope * (float)kpos) * inv_norm;
                if (kpos > qrow) v = -1e30f;  // causal mask
                st[t][r] = v;
            }
        }

        // ---- online softmax (row reductions over 16 lanes) ----
#pragma unroll
        for (int r = 0; r < 8; ++r) {
            float mx = fmaxf(st[0][r], st[1][r]);
#pragma unroll
            for (int off = 1; off < 16; off <<= 1)
                mx = fmaxf(mx, __shfl_xor(mx, off, 32));
            const float mnew = fmaxf(mrow[r], mx);
            const float corr = __expf(mrow[r] - mnew);
            const float p0 = __expf(st[0][r] - mnew);
            const float p1 = __expf(st[1][r] - mnew);
            float psum = p0 + p1;
#pragma unroll
            for (int off = 1; off < 16; off <<= 1)
                psum += __shfl_xor(psum, off, 32);
            lrow[r] = lrow[r] * corr + psum;
            mrow[r] = mnew;
            st[0][r] = p0;
            st[1][r] = p1;
#pragma unroll
            for (int c = 0; c < 4; ++c) o[c][r] *= corr;
        }

        // ---- P: C-layout -> A-layout via this wave's LDS region ----
#pragma unroll
        for (int t = 0; t < 2; ++t)
#pragma unroll
            for (int r = 0; r < 8; ++r)
                ps[(lg * 8 + r) * GLDS + t * 16 + ll] = (bf16)st[t][r];
        const bf16* pp = ps + ll * GLDS;
        v16bf pfr = ld_frag(pp + lg * 8, pp + 16 + lg * 8);

        // ---- O += P (16x32) @ V (32x64); V fragments from Vt, halves at lg*8 and 16+lg*8 ----
#pragma unroll
        for (int c = 0; c < 4; ++c) {
            const bf16* vp = Vp + (size_t)(c * 16 + ll) * S_LEN + kb + lg * 8;
            v16bf vf = ld_frag(vp, vp + 16);
            o[c] = __builtin_amdgcn_wmma_f32_16x16x32_bf16(
                false, pfr, false, vf, (short)0, o[c], false, false);
        }
    }

    // ---- normalize and store ctx (bf16) ----
#pragma unroll
    for (int c = 0; c < 4; ++c)
#pragma unroll
        for (int r = 0; r < 8; ++r) {
            const int row = q0 + lg * 8 + r;
            const int col = head * HDIM + c * 16 + ll;
            ctx[(size_t)row * D_DIM + col] = (bf16)(o[c][r] / lrow[r]);
        }
}

// ---------------------------------------------------------------------------
// Launch sequence.
// ---------------------------------------------------------------------------
extern "C" void kernel_launch(void* const* d_in, const int* in_sizes, int n_in,
                              void* d_out, int out_size, void* d_ws, size_t ws_size,
                              hipStream_t stream) {
    (void)in_sizes; (void)n_in; (void)out_size; (void)ws_size;

    const float* x     = (const float*)d_in[0];
    // d_in[1] = alibi, d_in[2] = attention_mask: computed analytically.
    const float* ln1_g = (const float*)d_in[3];
    const float* ln1_b = (const float*)d_in[4];
    const float* ln2_g = (const float*)d_in[5];
    const float* ln2_b = (const float*)d_in[6];
    const float* wqkv  = (const float*)d_in[7];
    const float* bqkv  = (const float*)d_in[8];
    const float* wo    = (const float*)d_in[9];
    const float* bo    = (const float*)d_in[10];
    const float* w1    = (const float*)d_in[11];
    const float* b1    = (const float*)d_in[12];
    const float* w2    = (const float*)d_in[13];
    const float* b2    = (const float*)d_in[14];
    float* out = (float*)d_out;

    // Workspace carve-up (bytes, 256-aligned chunks).
    char* base = (char*)d_ws;
    size_t off = 0;
    auto carve = [&](size_t bytes) {
        char* p = base + off;
        off += (bytes + 255) & ~(size_t)255;
        return p;
    };
    bf16* h      = (bf16*)carve((size_t)S_LEN * D_DIM * 2);
    bf16* qkvb   = (bf16*)carve((size_t)S_LEN * QKV_LD * 2);
    bf16* vt     = (bf16*)carve((size_t)D_DIM * S_LEN * 2);
    bf16* ctxb   = (bf16*)carve((size_t)S_LEN * D_DIM * 2);
    float* res   = (float*)carve((size_t)S_LEN * D_DIM * 4);
    bf16* h2     = (bf16*)carve((size_t)S_LEN * D_DIM * 2);
    bf16* mlp1   = (bf16*)carve((size_t)S_LEN * FF_DIM * 2);
    bf16* wqkvT  = (bf16*)carve((size_t)D_DIM * QKV_LD * 2);
    bf16* woT    = (bf16*)carve((size_t)D_DIM * D_DIM * 2);
    bf16* w1T    = (bf16*)carve((size_t)D_DIM * FF_DIM * 2);
    bf16* w2T    = (bf16*)carve((size_t)FF_DIM * D_DIM * 2);

    const dim3 tb(32, 8);
    // Weight transpose-convert (f32 [R,C] -> bf16 [C,R]).
    transpose_f32_bf16<<<dim3(QKV_LD / 32, D_DIM / 32), tb, 0, stream>>>(wqkv, wqkvT, D_DIM, QKV_LD);
    transpose_f32_bf16<<<dim3(D_DIM / 32, D_DIM / 32), tb, 0, stream>>>(wo, woT, D_DIM, D_DIM);
    transpose_f32_bf16<<<dim3(FF_DIM / 32, D_DIM / 32), tb, 0, stream>>>(w1, w1T, D_DIM, FF_DIM);
    transpose_f32_bf16<<<dim3(D_DIM / 32, FF_DIM / 32), tb, 0, stream>>>(w2, w2T, FF_DIM, D_DIM);

    // 1) input layernorm -> bf16
    ln_kernel<<<S_LEN, 256, 0, stream>>>(x, ln1_g, ln1_b, h, D_DIM);

    // 2) fused QKV (bf16 out)
    gemm_bf16_wmma<0, false, false, true>
        <<<dim3(QKV_LD / GBN, S_LEN / GBM), 256, 0, stream>>>(
        h, wqkvT, bqkv, nullptr, nullptr, qkvb, QKV_LD, D_DIM);

    // 3) per-head V transpose, then attention
    vtrans_kernel<<<dim3(S_LEN / 32, NHEAD * 2), tb, 0, stream>>>(qkvb, vt);
    attn_kernel<<<dim3(S_LEN / 64, NHEAD), 128, 0, stream>>>(qkvb, vt, ctxb);

    // 4) output projection + residual: res = x + ctx @ wo + bo  (f32 out)
    gemm_bf16_wmma<0, true, true, false>
        <<<dim3(D_DIM / GBN, S_LEN / GBM), 256, 0, stream>>>(
        ctxb, woT, bo, x, res, nullptr, D_DIM, D_DIM);

    // 5) post-attn layernorm -> bf16
    ln_kernel<<<S_LEN, 256, 0, stream>>>(res, ln2_g, ln2_b, h2, D_DIM);

    // 6) MLP up + exact GELU (bf16 out)
    gemm_bf16_wmma<1, false, false, true>
        <<<dim3(FF_DIM / GBN, S_LEN / GBM), 256, 0, stream>>>(
        h2, w1T, b1, nullptr, nullptr, mlp1, FF_DIM, D_DIM);

    // 7) MLP down + residual: out = res + mlp1 @ w2 + b2  (f32 out)
    gemm_bf16_wmma<0, true, true, false>
        <<<dim3(D_DIM / GBN, S_LEN / GBM), 256, 0, stream>>>(
        mlp1, w2T, b2, res, out, nullptr, D_DIM, FF_DIM);
}
